// GNNGroupTracker_79645873537769
// MI455X (gfx1250) — matrix operand order repaired and checked by reading, not compiled, not run on hardware
//
#include <hip/hip_runtime.h>
#include <hip/hip_bf16.h>
#include <math.h>

// ---------------- problem constants ----------------
constexpr int NN    = 50000;
constexpr int EE    = 500000;
constexpr int HID   = 96;
constexpr int HEADS = 4;
constexpr int DH    = 24;
constexpr int FMAP  = 32;
constexpr float EPSV = 1e-5f;

typedef __attribute__((ext_vector_type(16))) _Float16 v16h;
typedef __attribute__((ext_vector_type(8)))  float    v8f;

// ---------------- device helpers ----------------
__device__ __forceinline__ float gelu_f(float x) {
    return 0.5f * x * (1.0f + erff(x * 0.7071067811865475f));
}
__device__ __forceinline__ float softplus_f(float x) {
    return fmaxf(x, 0.0f) + log1pf(expf(-fabsf(x)));
}
// order-preserving float<->uint encoding for atomicMax on floats
__device__ __forceinline__ unsigned fenc(float f) {
    unsigned u = __float_as_uint(f);
    return (u & 0x80000000u) ? ~u : (u | 0x80000000u);
}
__device__ __forceinline__ float fdec(unsigned e) {
    unsigned u = (e & 0x80000000u) ? (e & 0x7fffffffu) : ~e;
    return __uint_as_float(u);
}
__device__ __forceinline__ v8f wmma_f16(v16h a, v16h b, v8f c) {
    return __builtin_amdgcn_wmma_f32_16x16x32_f16(false, a, false, b, (short)0, c, false, false);
}
// pack two 32B-aligned runs of 8 floats into an f16 A-fragment
__device__ __forceinline__ v16h packA(float4 p0, float4 p1, float4 p2, float4 p3) {
    v16h a;
    a[0]  = (_Float16)p0.x; a[1]  = (_Float16)p0.y; a[2]  = (_Float16)p0.z; a[3]  = (_Float16)p0.w;
    a[4]  = (_Float16)p1.x; a[5]  = (_Float16)p1.y; a[6]  = (_Float16)p1.z; a[7]  = (_Float16)p1.w;
    a[8]  = (_Float16)p2.x; a[9]  = (_Float16)p2.y; a[10] = (_Float16)p2.z; a[11] = (_Float16)p2.w;
    a[12] = (_Float16)p3.x; a[13] = (_Float16)p3.y; a[14] = (_Float16)p3.z; a[15] = (_Float16)p3.w;
    return a;
}

// ---------------- generic fill ----------------
__global__ void fill_f32(float* __restrict__ p, float v, int n) {
    int i = blockIdx.x * blockDim.x + threadIdx.x;
    if (i < n) p[i] = v;
}

// ---------------- WMMA GEMM: C[M x 96] = A[M x 96] @ B[96 x 96] (+bias) ----------------
// B staged in LDS pre-swizzled to per-lane fragment layout:
// frag f = (kchunk*6 + tile)*32 + lane ; element i <-> B[(kchunk*32 + (lane>=16?16:0) + i)*96 + tile*16 + (lane&15)]
__global__ __launch_bounds__(256) void gemm96(const float* __restrict__ A,
                                              const float* __restrict__ B,
                                              const float* __restrict__ bias,
                                              float* __restrict__ C, int M) {
    __shared__ _Float16 BsF[3 * 6 * 32 * 16]; // 9216 halves = 18 KB
    int tid = threadIdx.x;
    for (int f = tid; f < 3 * 6 * 32; f += 256) {
        int c = f / 192, rem = f % 192;
        int t = rem / 32, l = rem % 32;
        int n  = t * 16 + (l & 15);
        int kb = c * 32 + ((l >= 16) ? 16 : 0);
        v16h tmp;
#pragma unroll
        for (int i = 0; i < 16; i++) tmp[i] = (_Float16)B[(kb + i) * 96 + n];
        *(v16h*)&BsF[f * 16] = tmp;
    }
    __syncthreads();

    int lane = tid & 31, wave = tid >> 5;
    int m0 = blockIdx.x * 128 + wave * 16;
    int row = m0 + (lane & 15);
    if (row >= M) row = M - 1;               // clamp loads; stores guarded
    const float* Arow = A + (size_t)row * 96;

    v8f acc[6] = {};
    const int khiA = (lane >= 16) ? 16 : 0;

#pragma unroll
    for (int kc = 0; kc < 96; kc += 32) {
        const float* r0 = Arow + kc + khiA;       // 32B aligned
        const float* r1 = r0 + 8;
        float4 p0 = ((const float4*)r0)[0], p1 = ((const float4*)r0)[1];
        float4 p2 = ((const float4*)r1)[0], p3 = ((const float4*)r1)[1];
        v16h a = packA(p0, p1, p2, p3);
        int cbase = (kc / 32) * 6;
        // issue all 6 B-fragment LDS loads up front so WMMAs overlap with DS
        v16h bfr[6];
#pragma unroll
        for (int t = 0; t < 6; t++)
            bfr[t] = *(const v16h*)&BsF[((cbase + t) * 32 + lane) * 16];
#pragma unroll
        for (int t = 0; t < 6; t++)
            acc[t] = wmma_f16(a, bfr[t], acc[t]);
    }

    int rbase = m0 + ((lane >= 16) ? 8 : 0);
#pragma unroll
    for (int t = 0; t < 6; t++) {
        int n = t * 16 + (lane & 15);
        float bv = bias ? bias[n] : 0.0f;
#pragma unroll
        for (int r = 0; r < 8; r++) {
            int m = rbase + r;
            if (m < M) C[(size_t)m * 96 + n] = acc[t][r] + bv;
        }
    }
}

// ---------------- node encoder: fourier -> MLP(LN, gelu) -> h0 ----------------
__global__ void node_encode(const float* __restrict__ x, const float* __restrict__ fB,
                            const float* __restrict__ w1, const float* __restrict__ b1,
                            const float* __restrict__ g,  const float* __restrict__ be,
                            const float* __restrict__ w2, const float* __restrict__ b2,
                            float* __restrict__ h0) {
    __shared__ float xin[66];
    __shared__ float tb[96];
    __shared__ float red[2];
    int t = threadIdx.x;
    int node = blockIdx.x;
    if (t < 2) xin[t] = x[node * 2 + t];
    if (t < FMAP) {
        float x0 = x[node * 2], x1 = x[node * 2 + 1];
        float xp = 6.283185307179586f * (x0 * fB[t] + x1 * fB[FMAP + t]);
        xin[2 + t] = sinf(xp);
        xin[2 + FMAP + t] = cosf(xp);
    }
    __syncthreads();
    float u = b1[t];
    for (int k = 0; k < 66; k++) u += xin[k] * w1[k * 96 + t];
    tb[t] = u;
    __syncthreads();
    if (t == 0) {
        float s = 0.f, q = 0.f;
        for (int k = 0; k < 96; k++) { s += tb[k]; q += tb[k] * tb[k]; }
        float m = s / 96.f;
        red[0] = m;
        red[1] = rsqrtf(q / 96.f - m * m + EPSV);
    }
    __syncthreads();
    float tv = gelu_f((u - red[0]) * red[1] * g[t] + be[t]);
    __syncthreads();
    tb[t] = tv;
    __syncthreads();
    float o = b2[t];
    for (int k = 0; k < 96; k++) o += tb[k] * w2[k * 96 + t];
    h0[(size_t)node * 96 + t] = o;
}

// ---------------- edge encoder stage 1: T = gelu(LN(ea @ w1 + b1)) ----------------
__global__ void edge_pre(const float* __restrict__ ea, const float* __restrict__ w1,
                         const float* __restrict__ b1, const float* __restrict__ g,
                         const float* __restrict__ be, float* __restrict__ T) {
    __shared__ float u[96];
    __shared__ float red[2];
    int t = threadIdx.x;
    int eid = blockIdx.x;
    float a0 = ea[eid * 3], a1 = ea[eid * 3 + 1], a2 = ea[eid * 3 + 2];
    float uu = b1[t] + a0 * w1[t] + a1 * w1[96 + t] + a2 * w1[192 + t];
    u[t] = uu;
    __syncthreads();
    if (t == 0) {
        float s = 0.f, q = 0.f;
        for (int k = 0; k < 96; k++) { s += u[k]; q += u[k] * u[k]; }
        float m = s / 96.f;
        red[0] = m;
        red[1] = rsqrtf(q / 96.f - m * m + EPSV);
    }
    __syncthreads();
    T[(size_t)eid * 96 + t] = gelu_f((uu - red[0]) * red[1] * g[t] + be[t]);
}

// ---------------- attention pass 1: per (edge, head) logits + segment max ----------------
__global__ void edge_logits(const float* __restrict__ Q, const float* __restrict__ K,
                            const float* __restrict__ EW, const int* __restrict__ src,
                            const int* __restrict__ dst, float* __restrict__ logits,
                            unsigned* __restrict__ mEnc) {
    int idx = blockIdx.x * blockDim.x + threadIdx.x;
    if (idx >= EE * HEADS) return;
    int eid = idx >> 2, h = idx & 3;
    int sI = src[eid], dI = dst[eid];
    const float4* q  = (const float4*)(Q  + (size_t)dI * 96 + h * DH);
    const float4* k  = (const float4*)(K  + (size_t)sI * 96 + h * DH);
    const float4* ew = (const float4*)(EW + (size_t)eid * 96 + h * DH);
    float acc = 0.f;
#pragma unroll
    for (int i = 0; i < DH / 4; i++) {
        float4 qa = q[i], ka = k[i], ea = ew[i];
        acc += qa.x * (ka.x + ea.x) + qa.y * (ka.y + ea.y)
             + qa.z * (ka.z + ea.z) + qa.w * (ka.w + ea.w);
    }
    float lg = acc * 0.20412414523193154f; // 1/sqrt(24)
    logits[idx] = lg;
    atomicMax(&mEnc[dI * 4 + h], fenc(lg));
}

// ---------------- attention pass 2: accumulate sum(exp) and sum(exp * v) ----------------
__global__ void edge_accum(const float* __restrict__ logits, const unsigned* __restrict__ mEnc,
                           const float* __restrict__ V, const float* __restrict__ EW,
                           const int* __restrict__ src, const int* __restrict__ dst,
                           float* __restrict__ ssum, float* __restrict__ msg) {
    int idx = blockIdx.x * blockDim.x + threadIdx.x;
    if (idx >= EE * HEADS) return;
    int eid = idx >> 2, h = idx & 3;
    int sI = src[eid], dI = dst[eid];
    float m = fdec(mEnc[dI * 4 + h]);
    float ex = expf(logits[idx] - m);
    atomicAdd(&ssum[dI * 4 + h], ex);
    const float4* v  = (const float4*)(V  + (size_t)sI * 96 + h * DH);
    const float4* ew = (const float4*)(EW + (size_t)eid * 96 + h * DH);
    float* mp = msg + (size_t)dI * 96 + h * DH;
#pragma unroll
    for (int i = 0; i < DH / 4; i++) {
        float4 va = v[i], ea = ew[i];
        atomicAdd(&mp[i * 4 + 0], ex * (va.x + ea.x));
        atomicAdd(&mp[i * 4 + 1], ex * (va.y + ea.y));
        atomicAdd(&mp[i * 4 + 2], ex * (va.z + ea.z));
        atomicAdd(&mp[i * 4 + 3], ex * (va.w + ea.w));
    }
}

// ---------------- node update pre-BN: pre = msg/s + skip + bskip; BN partial sums ----------------
__global__ void node_pre(const float* __restrict__ msg, const float* __restrict__ ssum,
                         const float* __restrict__ SK, const float* __restrict__ bsk,
                         float* __restrict__ pre, float* __restrict__ bnSum,
                         float* __restrict__ bnSqr) {
    int c = threadIdx.x;          // 0..95 (channel)
    int nb = blockIdx.x * 64;
    float s = 0.f, q = 0.f;
    float bc = bsk[c];
    for (int j = 0; j < 64; j++) {
        int n = nb + j;
        if (n >= NN) break;
        float sv = ssum[n * 4 + c / DH];
        float val = (sv > 0.f ? msg[(size_t)n * 96 + c] / sv : 0.f)
                    + SK[(size_t)n * 96 + c] + bc;
        pre[(size_t)n * 96 + c] = val;
        s += val; q += val * val;
    }
    atomicAdd(&bnSum[c], s);
    atomicAdd(&bnSqr[c], q);
}

// ---------------- node update finalize: gelu(BN(pre)) + residual (float4) ----------------
__global__ void node_fin(const float* __restrict__ pre, const float* __restrict__ hin,
                         const float* __restrict__ bnSum, const float* __restrict__ bnSqr,
                         const float* __restrict__ g, const float* __restrict__ b,
                         float* __restrict__ hout) {
    int idx4 = blockIdx.x * blockDim.x + threadIdx.x;
    if (idx4 >= NN * (96 / 4)) return;
    int c0 = (idx4 * 4) % 96;
    float4 pv = ((const float4*)pre)[idx4];
    float4 hv = ((const float4*)hin)[idx4];
    float o[4]; float pvv[4] = {pv.x, pv.y, pv.z, pv.w};
    float hvv[4] = {hv.x, hv.y, hv.z, hv.w};
#pragma unroll
    for (int j = 0; j < 4; j++) {
        int c = c0 + j;
        float mean = bnSum[c] * (1.0f / NN);
        float var  = bnSqr[c] * (1.0f / NN) - mean * mean;
        o[j] = gelu_f((pvv[j] - mean) * rsqrtf(var + EPSV) * g[c] + b[c]) + hvv[j];
    }
    ((float4*)hout)[idx4] = make_float4(o[0], o[1], o[2], o[3]);
}

// ---------------- fusion weights: alpha = softmax(mean(fusion_attn, -1)) ----------------
__global__ void alpha_k(const float* __restrict__ fa, float* __restrict__ alpha) {
    __shared__ float m[4];
    int t = threadIdx.x;
    if (t < 4) {
        float s = 0.f;
        for (int k = 0; k < 96; k++) s += fa[t * 96 + k];
        m[t] = s / 96.f;
    }
    __syncthreads();
    if (t == 0) {
        float mx = fmaxf(fmaxf(m[0], m[1]), fmaxf(m[2], m[3]));
        float e0 = expf(m[0] - mx), e1 = expf(m[1] - mx),
              e2 = expf(m[2] - mx), e3 = expf(m[3] - mx);
        float s = e0 + e1 + e2 + e3;
        alpha[0] = e0 / s; alpha[1] = e1 / s; alpha[2] = e2 / s; alpha[3] = e3 / s;
    }
}

__global__ void fuse_k(const float* __restrict__ h1, const float* __restrict__ h2,
                       const float* __restrict__ h3, const float* __restrict__ h4,
                       const float* __restrict__ alpha, float* __restrict__ hf) {
    int idx4 = blockIdx.x * blockDim.x + threadIdx.x;
    if (idx4 >= NN * (96 / 4)) return;
    float a0 = alpha[0], a1 = alpha[1], a2 = alpha[2], a3 = alpha[3];
    float4 v1 = ((const float4*)h1)[idx4], v2 = ((const float4*)h2)[idx4];
    float4 v3 = ((const float4*)h3)[idx4], v4 = ((const float4*)h4)[idx4];
    float4 o;
    o.x = a0 * v1.x + a1 * v2.x + a2 * v3.x + a3 * v4.x;
    o.y = a0 * v1.y + a1 * v2.y + a2 * v3.y + a3 * v4.y;
    o.z = a0 * v1.z + a1 * v2.z + a2 * v3.z + a3 * v4.z;
    o.w = a0 * v1.w + a1 * v2.w + a2 * v3.w + a3 * v4.w;
    ((float4*)hf)[idx4] = o;
}

// ---------------- WMMA edge classifier: sigmoid(gelu([hf[s],hf[d],e] @ W1 + b1) @ w2 + b2) ----------------
__global__ __launch_bounds__(256) void edge_cls(const float* __restrict__ hf,
                                                const float* __restrict__ eF,
                                                const int* __restrict__ src,
                                                const int* __restrict__ dst,
                                                const float* __restrict__ W1,
                                                const float* __restrict__ b1,
                                                const float* __restrict__ w2,
                                                const float* __restrict__ b2,
                                                float* __restrict__ escore) {
    __shared__ _Float16 WsF[9 * 6 * 32 * 16]; // 27648 halves = 55296 B
    int tid = threadIdx.x;
    for (int f = tid; f < 9 * 6 * 32; f += 256) {
        int c = f / 192, rem = f % 192;
        int t = rem / 32, l = rem % 32;
        int n  = t * 16 + (l & 15);
        int kb = c * 32 + ((l >= 16) ? 16 : 0);
        v16h tmp;
#pragma unroll
        for (int i = 0; i < 16; i++) tmp[i] = (_Float16)W1[(kb + i) * 96 + n];
        *(v16h*)&WsF[f * 16] = tmp;
    }
    __syncthreads();

    int lane = tid & 31, wave = tid >> 5;
    int e0 = blockIdx.x * 128 + wave * 16;
    int erow = e0 + (lane & 15);
    if (erow >= EE) erow = EE - 1;
    const float* ps = hf + (size_t)src[erow] * 96;
    const float* pd = hf + (size_t)dst[erow] * 96;
    const float* pe = eF + (size_t)erow * 96;

    v8f acc[6] = {};
    const int khiA = (lane >= 16) ? 16 : 0;

#pragma unroll
    for (int kc = 0; kc < 288; kc += 32) {
        int k0 = kc + khiA;       // run of 8, never crosses a 96-float segment
        int k1 = k0 + 8;
        const float* r0 = (k0 < 96) ? ps + k0 : (k0 < 192 ? pd + k0 - 96 : pe + k0 - 192);
        const float* r1 = (k1 < 96) ? ps + k1 : (k1 < 192 ? pd + k1 - 96 : pe + k1 - 192);
        float4 p0 = ((const float4*)r0)[0], p1 = ((const float4*)r0)[1];
        float4 p2 = ((const float4*)r1)[0], p3 = ((const float4*)r1)[1];
        v16h a = packA(p0, p1, p2, p3);
        int cbase = (kc / 32) * 6;
        v16h bfr[6];
#pragma unroll
        for (int t = 0; t < 6; t++)
            bfr[t] = *(const v16h*)&WsF[((cbase + t) * 32 + lane) * 16];
#pragma unroll
        for (int t = 0; t < 6; t++)
            acc[t] = wmma_f16(a, bfr[t], acc[t]);
    }

    // epilogue: bias + gelu + project to scalar per edge, reduce across the 16-lane N groups
    float rsum[8];
#pragma unroll
    for (int r = 0; r < 8; r++) rsum[r] = 0.f;
#pragma unroll
    for (int t = 0; t < 6; t++) {
        int n = t * 16 + (lane & 15);
        float bb = b1[n], ww = w2[n];
#pragma unroll
        for (int r = 0; r < 8; r++) rsum[r] += gelu_f(acc[t][r] + bb) * ww;
    }
#pragma unroll
    for (int msk = 1; msk < 16; msk <<= 1) {
#pragma unroll
        for (int r = 0; r < 8; r++) rsum[r] += __shfl_xor(rsum[r], msk, 16);
    }
    if ((lane & 15) == 0) {
        int rb = e0 + ((lane >= 16) ? 8 : 0);
        float bias2 = b2[0];
#pragma unroll
        for (int r = 0; r < 8; r++) {
            int eidx = rb + r;
            if (eidx < EE) escore[eidx] = 1.f / (1.f + expf(-(rsum[r] + bias2)));
        }
    }
}

// ---------------- output regressor ----------------
__global__ void regress(const float* __restrict__ hf, const float* __restrict__ W1,
                        const float* __restrict__ b1, const float* __restrict__ W2,
                        const float* __restrict__ b2, float* __restrict__ offs,
                        float* __restrict__ unc) {
    __shared__ float hr[96];
    __shared__ float hid[96];
    int t = threadIdx.x;
    int n = blockIdx.x;
    hr[t] = hf[(size_t)n * 96 + t];
    __syncthreads();
    float u = b1[t];
    for (int k = 0; k < 96; k++) u += hr[k] * W1[k * 96 + t];
    hid[t] = gelu_f(u);
    __syncthreads();
    if (t < 4) {
        float o = b2[t];
        for (int k = 0; k < 96; k++) o += hid[k] * W2[k * 4 + t];
        if (t < 2) offs[n * 2 + t] = o;
        else       unc[n * 2 + (t - 2)] = softplus_f(o) + 1e-6f;
    }
}

// ---------------- host orchestration ----------------
extern "C" void kernel_launch(void* const* d_in, const int* in_sizes, int n_in,
                              void* d_out, int out_size, void* d_ws, size_t ws_size,
                              hipStream_t stream) {
    (void)in_sizes; (void)n_in; (void)out_size; (void)ws_size;
    const float* x    = (const float*)d_in[0];
    const int*   eidx = (const int*)d_in[1];
    const float* ea   = (const float*)d_in[2];
    const int* src = eidx;
    const int* dst = eidx + EE;

    auto P = [&](int i) { return (const float*)d_in[i]; };
    // params in setup_inputs() insertion order, starting at d_in[3]
    const float* fourier_B = P(3);
    const float* nm_w1 = P(4),  *nm_b1 = P(5),  *nm_g = P(6),  *nm_be = P(7);
    const float* nm_w2 = P(8),  *nm_b2 = P(9);
    const float* ee_w1 = P(10), *ee_b1 = P(11), *ee_g = P(12), *ee_be = P(13);
    const float* ee_w2 = P(14), *ee_b2 = P(15);
    const float* fusion_attn = P(16);
    const float* ec_w1 = P(17), *ec_b1 = P(18), *ec_w2 = P(19), *ec_b2 = P(20);
    const float* or_w1 = P(21), *or_b1 = P(22), *or_w2 = P(23), *or_b2 = P(24);
    // layer l (0..3): base 25 + l*11 : wq,bq,wk,bk,wv,bv,we,wskip,bskip,bn_g,bn_b

    // workspace layout (floats)
    float* ws = (float*)d_ws;
    size_t off = 0;
    auto alloc = [&](size_t nf) { float* p = ws + off; off += nf; return p; };
    float* hb[5];
    for (int i = 0; i < 5; i++) hb[i] = alloc((size_t)NN * 96);
    float* eB   = alloc((size_t)EE * 96);
    float* ewB  = alloc((size_t)EE * 96);   // also reused as edge-encoder temp T
    float* Qb   = alloc((size_t)NN * 96);   // also reused as BN "pre" buffer
    float* Kb   = alloc((size_t)NN * 96);
    float* Vb   = alloc((size_t)NN * 96);
    float* SKb  = alloc((size_t)NN * 96);
    float* lgB  = alloc((size_t)EE * 4);
    // contiguous zero region: msg | ssum | mEnc | bnSum | bnSqr
    float* msgB  = alloc((size_t)NN * 96);
    float* ssumB = alloc((size_t)NN * 4);
    unsigned* mEncB = (unsigned*)alloc((size_t)NN * 4);
    float* bnSum = alloc(96);
    float* bnSqr = alloc(96);
    const int zeroLen = NN * 96 + NN * 8 + 192;
    float* hfB    = alloc((size_t)NN * 96);
    float* alphaB = alloc(4);

    float* out_escore = (float*)d_out;
    float* out_offs   = out_escore + EE;
    float* out_unc    = out_offs + 2 * NN;

    const dim3 blk256(256);
    const int gN128 = (NN + 127) / 128;
    const int gE128 = (EE + 127) / 128;
    const int gEH   = (EE * HEADS + 255) / 256;
    const int gN4   = (NN * 24 + 255) / 256;

    // encoders
    node_encode<<<NN, 96, 0, stream>>>(x, fourier_B, nm_w1, nm_b1, nm_g, nm_be,
                                       nm_w2, nm_b2, hb[0]);
    edge_pre<<<EE, 96, 0, stream>>>(ea, ee_w1, ee_b1, ee_g, ee_be, ewB);
    gemm96<<<gE128, blk256, 0, stream>>>(ewB, ee_w2, ee_b2, eB, EE);

    // transformer-conv layers
    for (int l = 0; l < 4; l++) {
        int pb = 25 + l * 11;
        const float* wq = P(pb + 0), *bq  = P(pb + 1);
        const float* wk = P(pb + 2), *bk  = P(pb + 3);
        const float* wv = P(pb + 4), *bv  = P(pb + 5);
        const float* we = P(pb + 6);
        const float* wsk = P(pb + 7), *bsk = P(pb + 8);
        const float* bng = P(pb + 9), *bnb = P(pb + 10);
        const float* hin = hb[l];
        float* hout = hb[l + 1];

        fill_f32<<<(zeroLen + 255) / 256, blk256, 0, stream>>>(msgB, 0.0f, zeroLen);
        gemm96<<<gN128, blk256, 0, stream>>>(hin, wq, bq, Qb, NN);
        gemm96<<<gN128, blk256, 0, stream>>>(hin, wk, bk, Kb, NN);
        gemm96<<<gN128, blk256, 0, stream>>>(hin, wv, bv, Vb, NN);
        gemm96<<<gN128, blk256, 0, stream>>>(hin, wsk, nullptr, SKb, NN);
        gemm96<<<gE128, blk256, 0, stream>>>(eB, we, nullptr, ewB, EE);

        edge_logits<<<gEH, blk256, 0, stream>>>(Qb, Kb, ewB, src, dst, lgB, mEncB);
        edge_accum<<<gEH, blk256, 0, stream>>>(lgB, mEncB, Vb, ewB, src, dst, ssumB, msgB);
        node_pre<<<(NN + 63) / 64, 96, 0, stream>>>(msgB, ssumB, SKb, bsk, Qb, bnSum, bnSqr);
        node_fin<<<gN4, blk256, 0, stream>>>(Qb, hin, bnSum, bnSqr, bng, bnb, hout);
    }

    // fusion + heads
    alpha_k<<<1, 32, 0, stream>>>(fusion_attn, alphaB);
    fuse_k<<<gN4, blk256, 0, stream>>>(hb[1], hb[2], hb[3], hb[4], alphaB, hfB);
    edge_cls<<<gE128, blk256, 0, stream>>>(hfB, eB, src, dst, ec_w1, ec_b1, ec_w2, ec_b2,
                                           out_escore);
    regress<<<NN, 96, 0, stream>>>(hfB, or_w1, or_b1, or_w2, or_b2, out_offs, out_unc);
}
